// VectorQuantizer_90426241450828
// MI455X (gfx1250) — matrix operand order repaired and checked by reading, not compiled, not run on hardware
//
#include <hip/hip_runtime.h>
#include <hip/hip_bf16.h>

typedef __attribute__((ext_vector_type(16))) __bf16 v16bf;
typedef __attribute__((ext_vector_type(8)))  float  v8f;

#define NUM_EMB   1024
#define EMB_DIM   64
#define BROWS     128      // rows (flat vectors) per block
#define NTHREADS  256      // 8 waves of 32
#define HWSZ      4096     // 64*64 spatial per image
#define TOTAL_EL  4194304  // 16*64*64*64

// workspace byte offsets
#define WS_BHI   0
#define WS_BLO   (128*1024)
#define WS_CN    (256*1024)
#define WS_PART  (260*1024)

// ---------------------------------------------------------------------------
// Kernel 1: swizzle codebook into WMMA B-fragment order as bf16 hi/lo pair,
// and compute per-entry squared norms in fp32.
// B layout (bf16 16x16x32): lane%16 = column (codebook entry), lane/16 picks
// K-half; each lane holds 16 consecutive d-values. Fragment id = tile*2+chunk.
// ---------------------------------------------------------------------------
__global__ __launch_bounds__(256) void vq_prep(const float* __restrict__ cb,
                                               __bf16* __restrict__ bhi,
                                               __bf16* __restrict__ blo,
                                               float* __restrict__ cn) {
  int id = blockIdx.x * 256 + threadIdx.x;   // 0..4095 : (tile*2+chunk)*32 + lane
  if (id < 4096) {
    int lane = id & 31;
    int tc   = id >> 5;
    int c    = tc & 1;
    int t    = tc >> 1;
    int k    = t * 16 + (lane & 15);
    int dbase = c * 32 + ((lane >> 4) << 4);
    const float* s = cb + k * EMB_DIM + dbase;
    __bf16* dh = bhi + (size_t)id * 16;
    __bf16* dl = blo + (size_t)id * 16;
#pragma unroll
    for (int j = 0; j < 16; ++j) {
      float v = s[j];
      __bf16 h = (__bf16)v;
      dh[j] = h;
      dl[j] = (__bf16)(v - (float)h);
    }
  }
  if (id < NUM_EMB) {
    const float* s = cb + id * EMB_DIM;
    float acc = 0.f;
#pragma unroll
    for (int j = 0; j < EMB_DIM; ++j) acc += s[j] * s[j];
    cn[id] = acc;
  }
}

// ---------------------------------------------------------------------------
// Kernel 2: main VQ kernel. One block handles 128 flat rows (contiguous hw
// within one batch image). Scores via bf16 split-precision WMMA, argmin,
// gather, NCHW output, per-block loss partial.
// ---------------------------------------------------------------------------
__global__ __launch_bounds__(256) void vq_main(const float* __restrict__ x,
                                               const float* __restrict__ cb,
                                               const __bf16* __restrict__ bhi,
                                               const __bf16* __restrict__ blo,
                                               const float* __restrict__ cnorm,
                                               float* __restrict__ out,
                                               float* __restrict__ part) {
  __shared__ float xs[BROWS][68];   // fp32 x tile, padded (+4) against bank conflicts
  __shared__ float cn[NUM_EMB];
  __shared__ int   idxs[BROWS];
  __shared__ float red[NTHREADS];

  const int tid = threadIdx.x;
  const int bid = blockIdx.x;
  const int b   = bid >> 5;            // batch image
  const int hw0 = (bid & 31) << 7;     // first hw of this 128-row tile
  const float* xb = x + (size_t)b * (EMB_DIM * HWSZ) + hw0;

  // ---- load x tile (coalesced float4), store row-major fp32 into LDS ----
#pragma unroll
  for (int i = 0; i < 8; ++i) {
    int e  = i * 256 + tid;            // 2048 float4s
    int d  = e >> 5;
    int hw = (e & 31) << 2;
    float4 v = *(const float4*)(xb + (size_t)d * HWSZ + hw);
    xs[hw + 0][d] = v.x;
    xs[hw + 1][d] = v.y;
    xs[hw + 2][d] = v.z;
    xs[hw + 3][d] = v.w;
  }
#pragma unroll
  for (int i = 0; i < 4; ++i) cn[i * 256 + tid] = cnorm[i * 256 + tid];
  __syncthreads();

  const int lane = tid & 31;
  const int w    = tid >> 5;           // wave id: owns rows w*16 .. w*16+15
  const int nrow = lane & 15;
  const int half = lane >> 4;
  const int arow = w * 16 + nrow;

  // ---- build A fragments (hi/lo bf16 split), contiguous k order, same as B pack ----
  // lanes<16: K runs {0..15}; lanes>=16: {16..31} (per 32-chunk)
  v16bf Ah[2], Al[2];
#pragma unroll
  for (int c = 0; c < 2; ++c) {
    int base = c * 32 + (half << 4);
    float f[16];
    *(float4*)&f[0]  = *(const float4*)&xs[arow][base];
    *(float4*)&f[4]  = *(const float4*)&xs[arow][base + 4];
    *(float4*)&f[8]  = *(const float4*)&xs[arow][base + 8];
    *(float4*)&f[12] = *(const float4*)&xs[arow][base + 12];
#pragma unroll
    for (int j = 0; j < 16; ++j) {
      float v  = f[j];
      __bf16 h = (__bf16)v;
      Ah[c][j] = h;
      Al[c][j] = (__bf16)(v - (float)h);
    }
  }

  // ---- running argmin over the 64 column tiles ----
  float minv[8];
  int   mink[8];
#pragma unroll
  for (int r = 0; r < 8; ++r) { minv[r] = 3.402823466e38f; mink[r] = 0; }

  for (int t = 0; t < 64; ++t) {
    const __bf16* p0 = bhi + ((size_t)(t * 2 + 0) * 32 + lane) * 16;
    const __bf16* p1 = bhi + ((size_t)(t * 2 + 1) * 32 + lane) * 16;
    const __bf16* q0 = blo + ((size_t)(t * 2 + 0) * 32 + lane) * 16;
    const __bf16* q1 = blo + ((size_t)(t * 2 + 1) * 32 + lane) * 16;
    v16bf Bh0 = *(const v16bf*)p0;
    v16bf Bh1 = *(const v16bf*)p1;
    v16bf Bl0 = *(const v16bf*)q0;
    v16bf Bl1 = *(const v16bf*)q1;

    v8f acc = {};
    acc = __builtin_amdgcn_wmma_f32_16x16x32_bf16(false, Ah[0], false, Bh0, (short)0, acc, false, false);
    acc = __builtin_amdgcn_wmma_f32_16x16x32_bf16(false, Ah[1], false, Bh1, (short)0, acc, false, false);
    acc = __builtin_amdgcn_wmma_f32_16x16x32_bf16(false, Ah[0], false, Bl0, (short)0, acc, false, false);
    acc = __builtin_amdgcn_wmma_f32_16x16x32_bf16(false, Ah[1], false, Bl1, (short)0, acc, false, false);
    acc = __builtin_amdgcn_wmma_f32_16x16x32_bf16(false, Al[0], false, Bh0, (short)0, acc, false, false);
    acc = __builtin_amdgcn_wmma_f32_16x16x32_bf16(false, Al[1], false, Bh1, (short)0, acc, false, false);

    int   kcol = t * 16 + nrow;        // this lane's codebook column
    float c2   = cn[kcol];
#pragma unroll
    for (int r = 0; r < 8; ++r) {      // dist proxy: ||c||^2 - 2 x.c
      float dv = c2 - 2.0f * acc[r];
      if (dv < minv[r]) { minv[r] = dv; mink[r] = kcol; }  // tiles ascend -> first-min ties
    }
  }

  // ---- argmin reduction across the 16 columns held by each lane-half ----
#pragma unroll
  for (int off = 8; off >= 1; off >>= 1) {
#pragma unroll
    for (int r = 0; r < 8; ++r) {
      float ov = __shfl_xor(minv[r], off, 32);
      int   ok = __shfl_xor(mink[r], off, 32);
      if (ov < minv[r] || (ov == minv[r] && ok < mink[r])) { minv[r] = ov; mink[r] = ok; }
    }
  }
  if (nrow == 0) {                     // lane 0 -> rows r; lane 16 -> rows 8+r
#pragma unroll
    for (int r = 0; r < 8; ++r) idxs[w * 16 + half * 8 + r] = mink[r];
  }
  __syncthreads();

  // ---- gather + NCHW output (coalesced per d) + loss partial ----
  int row = tid & 127;
  int d0  = (tid >> 7) << 5;           // thread covers 32 d-values = one 128B codebook line
  int kk  = idxs[row];
  const float* crow = cb + (size_t)kk * EMB_DIM + d0;
  float* op = out + (size_t)b * (EMB_DIM * HWSZ) + (size_t)d0 * HWSZ + hw0 + row;
  float lsum = 0.f;
#pragma unroll
  for (int j = 0; j < 8; ++j) {
    float4 q = *(const float4*)(crow + 4 * j);
    float x0 = xs[row][d0 + 4 * j + 0];
    float x1 = xs[row][d0 + 4 * j + 1];
    float x2 = xs[row][d0 + 4 * j + 2];
    float x3 = xs[row][d0 + 4 * j + 3];
    float e0 = q.x - x0, e1 = q.y - x1, e2 = q.z - x2, e3 = q.w - x3;
    lsum += e0 * e0 + e1 * e1 + e2 * e2 + e3 * e3;
    op[(size_t)(4 * j + 0) * HWSZ] = q.x;
    op[(size_t)(4 * j + 1) * HWSZ] = q.y;
    op[(size_t)(4 * j + 2) * HWSZ] = q.z;
    op[(size_t)(4 * j + 3) * HWSZ] = q.w;
  }
  red[tid] = lsum;
  __syncthreads();
  for (int s = 128; s > 0; s >>= 1) {
    if (tid < s) red[tid] += red[tid + s];
    __syncthreads();
  }
  if (tid == 0) part[bid] = red[0];
}

// ---------------------------------------------------------------------------
// Kernel 3: deterministic final loss reduction.
// loss = (1 + 0.25) * mean((q - x)^2)   (both loss terms equal in value)
// ---------------------------------------------------------------------------
__global__ __launch_bounds__(256) void vq_loss(const float* __restrict__ part,
                                               float* __restrict__ out) {
  __shared__ float red[256];
  int tid = threadIdx.x;
  red[tid] = part[tid] + part[tid + 256];
  __syncthreads();
  for (int s = 128; s > 0; s >>= 1) {
    if (tid < s) red[tid] += red[tid + s];
    __syncthreads();
  }
  if (tid == 0) out[TOTAL_EL] = 1.25f * red[0] / (float)TOTAL_EL;
}

extern "C" void kernel_launch(void* const* d_in, const int* in_sizes, int n_in,
                              void* d_out, int out_size, void* d_ws, size_t ws_size,
                              hipStream_t stream) {
  const float* x  = (const float*)d_in[0];  // [16,64,64,64] fp32 NCHW
  const float* cb = (const float*)d_in[1];  // [1024,64] fp32
  float* out = (float*)d_out;               // 4194304 quantized + 1 loss

  char* ws = (char*)d_ws;
  __bf16* bhi  = (__bf16*)(ws + WS_BHI);    // 128KB swizzled hi codebook
  __bf16* blo  = (__bf16*)(ws + WS_BLO);    // 128KB swizzled lo codebook
  float*  cn   = (float*)(ws + WS_CN);      // 4KB ||c_k||^2
  float*  prt  = (float*)(ws + WS_PART);    // 2KB block loss partials

  vq_prep<<<16, 256, 0, stream>>>(cb, bhi, blo, cn);
  vq_main<<<512, 256, 0, stream>>>(x, cb, bhi, blo, cn, out, prt);
  vq_loss<<<1, 256, 0, stream>>>(prt, out);
}
